// InfoNCELoss_52312701665356
// MI455X (gfx1250) — matrix operand order repaired and checked by reading, not compiled, not run on hardware
//
#include <hip/hip_runtime.h>
#include <hip/hip_bf16.h>
#include <hip/hip_fp16.h>

typedef _Float16 half8  __attribute__((ext_vector_type(8)));
typedef _Float16 half16 __attribute__((ext_vector_type(16)));
typedef float    v8f    __attribute__((ext_vector_type(8)));

#define DIMK    512
#define BROWS   4096
#define NROWS   8192
#define NTILES  512            // NROWS / 16
#define INV_T   14.2857142857142857f  // 1 / 0.07

// ---------------------------------------------------------------------------
// Kernel 1: L2-normalize rows of [z_i; z_j] and emit f16 matrix z_h[8192][512]
// One wave32 per row; lane handles 16 contiguous floats.
// ---------------------------------------------------------------------------
__global__ __launch_bounds__(256) void nce_normalize_kernel(
    const float* __restrict__ zi, const float* __restrict__ zj,
    _Float16* __restrict__ zh) {
  const int wave = threadIdx.x >> 5;
  const int lane = threadIdx.x & 31;
  const int row  = blockIdx.x * 8 + wave;          // 0..8191
  const float* src = (row < BROWS) ? (zi + (size_t)row * DIMK)
                                   : (zj + (size_t)(row - BROWS) * DIMK);
  const float4* s4 = (const float4*)(src + lane * 16);
  float4 v0 = s4[0], v1 = s4[1], v2 = s4[2], v3 = s4[3];

  float ss = v0.x*v0.x + v0.y*v0.y + v0.z*v0.z + v0.w*v0.w
           + v1.x*v1.x + v1.y*v1.y + v1.z*v1.z + v1.w*v1.w
           + v2.x*v2.x + v2.y*v2.y + v2.z*v2.z + v2.w*v2.w
           + v3.x*v3.x + v3.y*v3.y + v3.z*v3.z + v3.w*v3.w;
  #pragma unroll
  for (int off = 16; off > 0; off >>= 1) ss += __shfl_xor(ss, off, 32);

  const float inv = 1.0f / fmaxf(sqrtf(ss), 1e-12f);

  _Float16 h[16];
  h[ 0]=(_Float16)(v0.x*inv); h[ 1]=(_Float16)(v0.y*inv);
  h[ 2]=(_Float16)(v0.z*inv); h[ 3]=(_Float16)(v0.w*inv);
  h[ 4]=(_Float16)(v1.x*inv); h[ 5]=(_Float16)(v1.y*inv);
  h[ 6]=(_Float16)(v1.z*inv); h[ 7]=(_Float16)(v1.w*inv);
  h[ 8]=(_Float16)(v2.x*inv); h[ 9]=(_Float16)(v2.y*inv);
  h[10]=(_Float16)(v2.z*inv); h[11]=(_Float16)(v2.w*inv);
  h[12]=(_Float16)(v3.x*inv); h[13]=(_Float16)(v3.y*inv);
  h[14]=(_Float16)(v3.z*inv); h[15]=(_Float16)(v3.w*inv);

  half8* dst = (half8*)(zh + (size_t)row * DIMK + lane * 16);
  dst[0] = *(half8*)&h[0];
  dst[1] = *(half8*)&h[8];
}

// ---------------------------------------------------------------------------
// Kernel 2: fused sim = Z Z^T / T  +  per-row exp-sum (self excluded) + pos.
// Block = 8 waves, owns 16-row tile rt; waves split the 512 column tiles.
// A tile cached in registers with CDNA5 16-bit A VGPR layout; B loaded
// contiguously per the 16-bit B striping; 16x v_wmma_f32_16x16x32_f16 per
// column tile with two accumulators to break the D->C chain.
// ---------------------------------------------------------------------------
__global__ __launch_bounds__(256) void nce_simlse_kernel(
    const _Float16* __restrict__ zh, float* __restrict__ partial) {
  const int rt   = blockIdx.x;          // row tile 0..511
  const int wave = threadIdx.x >> 5;
  const int lane = threadIdx.x & 31;
  const int hi   = lane >> 4;           // lane half
  const int l15  = lane & 15;

  // ---- cache A tile (16 rows x 512) in registers, WMMA A layout ----
  // a[h] (h=0..15): K = k0 + (h<8 ? 0 : 16) + 8*hi + (h & 7)
  const _Float16* arow = zh + (size_t)(rt * 16 + l15) * DIMK + 8 * hi;
  half16 areg[16];
  #pragma unroll
  for (int kc = 0; kc < 16; ++kc) {
    union { half16 h16; half8 h8[2]; } u;
    u.h8[0] = *(const half8*)(arow + kc * 32);
    u.h8[1] = *(const half8*)(arow + kc * 32 + 16);
    areg[kc] = u.h16;
  }

  const int pt = (rt + 256) & 511;      // column tile holding the positives
  float S[8], P[8];
  #pragma unroll
  for (int r = 0; r < 8; ++r) { S[r] = 0.0f; P[r] = 0.0f; }

  for (int ct = wave; ct < NTILES; ct += 8) {
    // B layout: b[h] = Z[ct*16 + l15][k0 + (h<8 ? 0 : 16) + 8*hi + (h & 7)]
    const _Float16* brow = zh + (size_t)(ct * 16 + l15) * DIMK + 8 * hi;
    v8f c0 = {}, c1 = {};
    #pragma unroll
    for (int kc = 0; kc < 16; kc += 2) {
      union { half16 h16; half8 h8[2]; } b0, b1;
      b0.h8[0] = *(const half8*)(brow + kc * 32);
      b0.h8[1] = *(const half8*)(brow + kc * 32 + 16);
      b1.h8[0] = *(const half8*)(brow + kc * 32 + 32);
      b1.h8[1] = *(const half8*)(brow + kc * 32 + 48);
      c0 = __builtin_amdgcn_wmma_f32_16x16x32_f16(
               false, areg[kc],     false, b0.h16, (short)0, c0, false, false);
      c1 = __builtin_amdgcn_wmma_f32_16x16x32_f16(
               false, areg[kc + 1], false, b1.h16, (short)0, c1, false, false);
    }
    const bool dt = (ct == rt);         // self tile -> mask its diagonal
    const bool pp = (ct == pt);         // positive tile -> grab its diagonal
    #pragma unroll
    for (int r = 0; r < 8; ++r) {
      // C layout: elem r of lane -> row 8*hi + r, col l15
      const bool mydiag = (l15 == (8 * hi + r));
      const float val = (c0[r] + c1[r]) * INV_T;
      if (pp && mydiag) P[r] = val;     // positive IS in the denominator
      float e = __expf(val);
      if (dt && mydiag) e = 0.0f;       // exclude self-similarity
      S[r] += e;
    }
  }

  // ---- reduce the 16 lanes of each half (columns of this wave's tiles) ----
  #pragma unroll
  for (int r = 0; r < 8; ++r) {
    #pragma unroll
    for (int off = 1; off < 16; off <<= 1) {
      S[r] += __shfl_xor(S[r], off, 32);
      P[r] += __shfl_xor(P[r], off, 32);
    }
  }

  __shared__ float redS[8][16];
  __shared__ float redP[8][16];
  __shared__ float contrib[16];
  if (l15 == 0) {                       // lanes 0 and 16
    #pragma unroll
    for (int r = 0; r < 8; ++r) {
      redS[wave][8 * hi + r] = S[r];
      redP[wave][8 * hi + r] = P[r];
    }
  }
  __syncthreads();
  if (threadIdx.x < 16) {
    float s = 0.0f, p = 0.0f;
    #pragma unroll
    for (int w = 0; w < 8; ++w) { s += redS[w][threadIdx.x]; p += redP[w][threadIdx.x]; }
    contrib[threadIdx.x] = __logf(s) - p;   // lse - pos for this row
  }
  __syncthreads();
  if (threadIdx.x == 0) {
    float t = 0.0f;
    #pragma unroll
    for (int i = 0; i < 16; ++i) t += contrib[i];
    partial[rt] = t;
  }
}

// ---------------------------------------------------------------------------
// Kernel 3: sum of the 512 block partials / 8192 (row mean) -> d_out[0] (overwrite, no atomics)
// ---------------------------------------------------------------------------
__global__ __launch_bounds__(256) void nce_finalize_kernel(
    const float* __restrict__ partial, float* __restrict__ out) {
  __shared__ float buf[256];
  buf[threadIdx.x] = partial[threadIdx.x] + partial[threadIdx.x + 256];
  __syncthreads();
  #pragma unroll
  for (int s = 128; s > 0; s >>= 1) {
    if (threadIdx.x < s) buf[threadIdx.x] += buf[threadIdx.x + s];
    __syncthreads();
  }
  if (threadIdx.x == 0) out[0] = buf[0] * (1.0f / (float)NROWS);
}

extern "C" void kernel_launch(void* const* d_in, const int* in_sizes, int n_in,
                              void* d_out, int out_size, void* d_ws, size_t ws_size,
                              hipStream_t stream) {
  (void)in_sizes; (void)n_in; (void)out_size; (void)ws_size;
  const float* zi = (const float*)d_in[0];
  const float* zj = (const float*)d_in[1];

  _Float16* zh = (_Float16*)d_ws;                                  // 8 MB
  float* partial = (float*)((char*)d_ws + (size_t)NROWS * DIMK * sizeof(_Float16));

  nce_normalize_kernel<<<NROWS / 8, 256, 0, stream>>>(zi, zj, zh);
  nce_simlse_kernel<<<NTILES, 256, 0, stream>>>(zh, partial);
  nce_finalize_kernel<<<1, 256, 0, stream>>>(partial, (float*)d_out);
}
